// adain_23235773071762
// MI455X (gfx1250) — matrix operand order repaired and checked by reading, not compiled, not run on hardware
//
#include <hip/hip_runtime.h>

// ---------------------------------------------------------------------------
// CDNA5 (gfx1250) style-transfer forward pass.
// Heavy math: v_wmma_f32_16x16x32_bf16 (wave32), ds_load_b128 fragment loads,
// double-buffered LDS staging; attention part3 GEMM stages its B operand with
// global_load_async_to_lds_b128 (ASYNCcnt) from a bf16 softmax output.
// Mode flags are template parameters so each GEMM variant compiles to a lean
// specialized inner loop.
// ---------------------------------------------------------------------------

typedef __bf16 v16bf __attribute__((ext_vector_type(16)));
typedef __bf16 v8bf  __attribute__((ext_vector_type(8)));
typedef __bf16 v2bf  __attribute__((ext_vector_type(2)));
typedef float  v8f   __attribute__((ext_vector_type(8)));

__device__ __forceinline__ unsigned pack_bf2(float a, float b) {
#if __has_builtin(__builtin_amdgcn_cvt_pk_bf16_f32)
    auto pk = __builtin_amdgcn_cvt_pk_bf16_f32(a, b);
    return __builtin_bit_cast(unsigned, pk);
#else
    v2bf v; v[0] = (__bf16)a; v[1] = (__bf16)b;   // fptrunc = RNE
    return __builtin_bit_cast(unsigned, v);
#endif
}
__device__ __forceinline__ unsigned short f2bf_raw(float f) {
    __bf16 h = (__bf16)f;
    return __builtin_bit_cast(unsigned short, h);
}
__device__ __forceinline__ void wait_async0() {
#if __has_builtin(__builtin_amdgcn_s_wait_asynccnt)
    __builtin_amdgcn_s_wait_asynccnt(0);
#else
    asm volatile("s_wait_asynccnt 0" ::: "memory");
#endif
}

#define BM 64
#define BN 64
#define BKK 32
#define TPB 128
#define ASTR 40   // elements; 80B row stride -> 16B aligned, odd-dword bank spread
#define BSTR 40

__device__ __forceinline__ v16bf load_fragA(const unsigned short* As, int arow, int half) {
    v8bf a0 = *(const v8bf*)(As + arow * ASTR + 8 * half);
    v8bf a1 = *(const v8bf*)(As + arow * ASTR + 16 + 8 * half);
    return __builtin_shufflevector(a0, a1, 0,1,2,3,4,5,6,7,8,9,10,11,12,13,14,15);
}
__device__ __forceinline__ v16bf load_fragB(const unsigned short* Bs, int bcol, int half) {
    v8bf b0 = *(const v8bf*)(Bs + bcol * BSTR + 8 * half);
    v8bf b1 = *(const v8bf*)(Bs + bcol * BSTR + 16 + 8 * half);
    return __builtin_shufflevector(b0, b1, 0,1,2,3,4,5,6,7,8,9,10,11,12,13,14,15);
}
#define WMMA_BF16(A, B, C) \
    __builtin_amdgcn_wmma_f32_16x16x32_bf16(false, (A), false, (B), (short)0, (C), false, false)

// Batched GEMM: Y[b] = alpha * op(A[b]) * op(B[b]) (+bias)(+relu)(+resid)
// CONV3: im2col gather on the fly (k->(c,dy,dx), n->(oi,oj)), optional per-channel
//        mean subtraction (builds style - mean lazily).
// ASYNC16: B operand is bf16 stored [N][K] (row-contiguous in k); B tiles staged
//        with global_load_async_to_lds_b128 + s_wait_asynccnt.
template<int TRANSA, int TRANSB, int CONV3, int ASYNC16>
__global__ __launch_bounds__(TPB)
void gemm_wmma_kernel(const float* __restrict__ A, long long aBatch, int lda,
                      const float* __restrict__ Bm, long long bBatch, int ldb,
                      int Cin, int IH, int IW, int OW,
                      const float* __restrict__ subMean,
                      const float* __restrict__ bias,
                      float alpha, int doRelu,
                      const float* __restrict__ resid, long long rBatch,
                      float* __restrict__ Y, long long yBatch, int ldy,
                      int M, int N, int K)
{
    // double-buffered bf16 tiles: A as [m][k], B transposed as [n][k]
    __shared__ __align__(16) unsigned short AsmU[2][BM * ASTR];
    __shared__ __align__(16) unsigned short BsmU[2][BN * BSTR];

    const int batch = blockIdx.z;
    const int tileM = blockIdx.y * BM;
    const int tileN = blockIdx.x * BN;
    const int tid   = threadIdx.x;
    const int lane  = tid & 31;
    const int wave  = tid >> 5;
    const int r     = lane & 15;
    const int half  = lane >> 4;

    const float* Ab = A  + (size_t)batch * aBatch;
    const float* Bb = Bm + (size_t)batch * bBatch;
    const unsigned short* B16b = (const unsigned short*)Bm + (size_t)batch * bBatch;
    const float* mb = subMean ? (subMean + (size_t)batch * Cin) : nullptr;

    v8f acc0 = {}, acc1 = {}, acc2 = {}, acc3 = {};
    float2 aReg[8], bReg[8];

    const int nK = (K + BKK - 1) / BKK;

    // ---- staging helpers ----
    auto stageA = [&](int k0) {
        #pragma unroll
        for (int t = 0; t < 8; ++t) {
            int i  = tid + t * TPB;              // over BM*BKK/2 pairs
            int mm = i >> 4, kk = (i & 15) << 1;
            int gm = tileM + mm, gk = k0 + kk;
            float x0 = 0.f, x1 = 0.f;
            if (gm < M) {
                if (TRANSA) {
                    if (gk < K)     x0 = Ab[(size_t)gk * lda + gm];
                    if (gk + 1 < K) x1 = Ab[(size_t)(gk + 1) * lda + gm];
                } else {
                    if (gk < K)     x0 = Ab[(size_t)gm * lda + gk];
                    if (gk + 1 < K) x1 = Ab[(size_t)gm * lda + gk + 1];
                }
            }
            aReg[t].x = x0; aReg[t].y = x1;
        }
    };
    auto stageB = [&](int k0) {
        #pragma unroll
        for (int t = 0; t < 8; ++t) {
            int i = tid + t * TPB;               // over BN*BKK/2 pairs
            int nn, kk;
            if (TRANSB) { kk = (i & 15) << 1; nn = i >> 4; }
            else        { nn = i & 63;        kk = (i >> 6) << 1; }
            int gn = tileN + nn, gk = k0 + kk;
            float x0 = 0.f, x1 = 0.f;
            if (gn < N) {
                if (CONV3) {
                    int oi = gn / OW, oj = gn % OW;
                    #pragma unroll
                    for (int q = 0; q < 2; ++q) {
                        int g = gk + q;
                        if (g < K) {
                            int c = g / 9, r9 = g % 9;
                            int dy = r9 / 3, dx = r9 % 3;
                            float v = Bb[(size_t)c * (IH * IW) + (size_t)(oi + dy) * IW + (oj + dx)];
                            if (mb) v -= mb[c];
                            if (q == 0) x0 = v; else x1 = v;
                        }
                    }
                } else if (TRANSB) {
                    if (gk < K)     x0 = Bb[(size_t)gn * ldb + gk];
                    if (gk + 1 < K) x1 = Bb[(size_t)gn * ldb + gk + 1];
                } else {
                    if (gk < K)     x0 = Bb[(size_t)gk * ldb + gn];
                    if (gk + 1 < K) x1 = Bb[(size_t)(gk + 1) * ldb + gn];
                }
            }
            bReg[t].x = x0; bReg[t].y = x1;
        }
    };
    auto storeA = [&](int buf) {
        unsigned short* As = AsmU[buf];
        #pragma unroll
        for (int t = 0; t < 8; ++t) {
            int i  = tid + t * TPB;
            int mm = i >> 4, kk = (i & 15) << 1;
            *(unsigned*)(As + mm * ASTR + kk) = pack_bf2(aReg[t].x, aReg[t].y);
        }
    };
    auto storeB = [&](int buf) {
        unsigned short* Bs = BsmU[buf];
        #pragma unroll
        for (int t = 0; t < 8; ++t) {
            int i = tid + t * TPB;
            int nn, kk;
            if (TRANSB) { kk = (i & 15) << 1; nn = i >> 4; }
            else        { nn = i & 63;        kk = (i >> 6) << 1; }
            *(unsigned*)(Bs + nn * BSTR + kk) = pack_bf2(bReg[t].x, bReg[t].y);
        }
    };
    // async bf16 staging: one 16B chunk per lane, 256 chunks cover the tile
    auto asyncB = [&](int k0, int buf) {
        unsigned short* Bs = BsmU[buf];
        #pragma unroll
        for (int t = 0; t < 2; ++t) {
            int id = tid + t * TPB;              // 0..255
            int nn = id >> 2, q = id & 3;        // row, 16B chunk in row
            int gn = tileN + nn;
            int gk = k0 + q * 8;
            unsigned short* lp = Bs + nn * BSTR + q * 8;
            if (gn < N && gk + 8 <= K) {
                unsigned lds = (unsigned)(unsigned long long)lp;   // low 32 = LDS offset
                unsigned long long ga =
                    (unsigned long long)(B16b + (size_t)gn * ldb + gk);
                asm volatile("global_load_async_to_lds_b128 %0, %1, off"
                             :: "v"(lds), "v"(ga) : "memory");
            } else {
                v8bf z = {};
                *(v8bf*)lp = z;
            }
        }
    };

    // ---- prologue: fill buffer 0 ----
    if (ASYNC16) asyncB(0, 0); else stageB(0);
    stageA(0);
    storeA(0);
    if (ASYNC16) wait_async0(); else storeB(0);
    __syncthreads();

    const int arow = wave * 16 + r;
    for (int kt = 0; kt < nK; ++kt) {
        const int buf = kt & 1;
        const bool more = (kt + 1) < nK;
        if (more) {                       // issue next tile's loads early
            stageA((kt + 1) * BKK);
            if (ASYNC16) {
                asyncB((kt + 1) * BKK, buf ^ 1);
            } else {
                stageB((kt + 1) * BKK);
                if (!CONV3 && (kt + 2) < nK) {
                    int gk = (kt + 2) * BKK + (tid >> 2); if (gk > K - 1) gk = K - 1;
                    int gn = tileN + ((tid & 3) << 4);    if (gn > N - 1) gn = N - 1;
                    const float* p = TRANSB ? &Bb[(size_t)gn * ldb + gk]
                                            : &Bb[(size_t)gk * ldb + gn];
                    __builtin_prefetch(p, 0, 3);
                }
            }
        }

        // ---- compute: issue all fragment loads, then 4 back-to-back WMMAs ----
        const unsigned short* As = AsmU[buf];
        const unsigned short* Bs = BsmU[buf];
        v16bf af  = load_fragA(As, arow, half);
        v16bf bf0 = load_fragB(Bs, r,      half);
        v16bf bf1 = load_fragB(Bs, 16 + r, half);
        v16bf bf2 = load_fragB(Bs, 32 + r, half);
        v16bf bf3 = load_fragB(Bs, 48 + r, half);
        acc0 = WMMA_BF16(af, bf0, acc0);
        acc1 = WMMA_BF16(af, bf1, acc1);
        acc2 = WMMA_BF16(af, bf2, acc2);
        acc3 = WMMA_BF16(af, bf3, acc3);

        if (more) {
            storeA(buf ^ 1);
            if (!ASYNC16) storeB(buf ^ 1);
            else          wait_async0();      // next buffer's async tiles landed
        }
        __syncthreads();
    }

    // ---- epilogue: C/D layout row = e + 8*half, col = lane&15 ----
    const float* Rb = resid ? (resid + (size_t)batch * rBatch) : nullptr;
    float* Yb = Y + (size_t)batch * yBatch;
    #pragma unroll
    for (int j = 0; j < 4; ++j) {
        v8f a = (j == 0) ? acc0 : (j == 1) ? acc1 : (j == 2) ? acc2 : acc3;
        const int col = tileN + j * 16 + r;
        if (col < N) {
            #pragma unroll
            for (int e = 0; e < 8; ++e) {
                const int row = tileM + wave * 16 + e + 8 * half;
                if (row < M) {
                    float v = alpha * a[e];
                    if (bias)   v += bias[row];
                    if (doRelu) v = fmaxf(v, 0.f);
                    size_t off = (size_t)row * ldy + col;
                    if (Rb) v += Rb[off];
                    Yb[off] = v;
                }
            }
        }
    }
}

// per-(b,c) spatial mean/var normalization; unbiased variance (ddof=1)
__global__ __launch_bounds__(256)
void mvn_kernel(const float* __restrict__ x, float* __restrict__ xn,
                float* __restrict__ meanOut, int HW, float eps)
{
    __shared__ float s1[256], s2[256];
    const size_t base = (size_t)blockIdx.x * HW;
    float a = 0.f, b = 0.f;
    for (int i = threadIdx.x; i < HW; i += 256) {
        float v = x[base + i]; a += v; b += v * v;
    }
    s1[threadIdx.x] = a; s2[threadIdx.x] = b;
    __syncthreads();
    for (int s = 128; s > 0; s >>= 1) {
        if (threadIdx.x < s) { s1[threadIdx.x] += s1[threadIdx.x + s]; s2[threadIdx.x] += s2[threadIdx.x + s]; }
        __syncthreads();
    }
    const float mean = s1[0] / HW;
    const float var  = (s2[0] - (float)HW * mean * mean) / (float)(HW - 1);
    const float inv  = rsqrtf(var + eps);
    if (meanOut && threadIdx.x == 0) meanOut[blockIdx.x] = mean;
    for (int i = threadIdx.x; i < HW; i += 256)
        xn[base + i] = (x[base + i] - mean) * inv;
}

// FC_S statistic: out[c*B + b] = sum(g^2)/sum(g) over HW  (transposed layout)
__global__ __launch_bounds__(256)
void mcc_stat_kernel(const float* __restrict__ g, float* __restrict__ out,
                     int Bn, int C, int HW)
{
    __shared__ float s1[256], s2[256];
    const int b = blockIdx.x / C, c = blockIdx.x % C;
    const size_t base = (size_t)blockIdx.x * HW;
    float a = 0.f, q = 0.f;
    for (int i = threadIdx.x; i < HW; i += 256) {
        float v = g[base + i]; a += v; q += v * v;
    }
    s1[threadIdx.x] = a; s2[threadIdx.x] = q;
    __syncthreads();
    for (int s = 128; s > 0; s >>= 1) {
        if (threadIdx.x < s) { s1[threadIdx.x] += s1[threadIdx.x + s]; s2[threadIdx.x] += s2[threadIdx.x + s]; }
        __syncthreads();
    }
    if (threadIdx.x == 0) out[(size_t)c * Bn + b] = s2[0] / s1[0];
}

// in-place row softmax over length L; emits normalized row as bf16
__global__ __launch_bounds__(256)
void softmax_kernel(float* __restrict__ S, unsigned short* __restrict__ out16, int L)
{
    __shared__ float red[256];
    const size_t base = (size_t)blockIdx.x * L;
    float m = -3.402823466e38f;
    for (int i = threadIdx.x; i < L; i += 256) m = fmaxf(m, S[base + i]);
    red[threadIdx.x] = m; __syncthreads();
    for (int s = 128; s > 0; s >>= 1) {
        if (threadIdx.x < s) red[threadIdx.x] = fmaxf(red[threadIdx.x], red[threadIdx.x + s]);
        __syncthreads();
    }
    m = red[0]; __syncthreads();
    float sum = 0.f;
    for (int i = threadIdx.x; i < L; i += 256) {
        float e = expf(S[base + i] - m); S[base + i] = e; sum += e;
    }
    red[threadIdx.x] = sum; __syncthreads();
    for (int s = 128; s > 0; s >>= 1) {
        if (threadIdx.x < s) red[threadIdx.x] += red[threadIdx.x + s];
        __syncthreads();
    }
    const float inv = 1.f / red[0];
    for (int i = threadIdx.x; i < L; i += 256)
        out16[base + i] = f2bf_raw(S[base + i] * inv);
}

// p = safe_cbrt(relu(FFc*FCS[c,b] * part2 * part3))
__global__ __launch_bounds__(256)
void combine_kernel(const float* __restrict__ FFc, const float* __restrict__ fcs,
                    const float* __restrict__ p2, const float* __restrict__ p3,
                    float* __restrict__ p, int Bn, int C, int HW, long long total)
{
    long long idx = (long long)blockIdx.x * blockDim.x + threadIdx.x;
    if (idx >= total) return;
    const int c = (int)((idx / HW) % C);
    const int b = (int)(idx / ((long long)HW * C));
    const float part1 = FFc[idx] * fcs[(size_t)c * Bn + b];
    const float v = part1 * p2[idx] * p3[idx];
    p[idx] = (v > 0.f) ? powf(v, 1.0f / 3.0f) : 0.f;
}

template<int TA, int TB, int C3, int AS>
static void launch_gemm(hipStream_t s,
                        const float* A, long long aB, int lda,
                        const float* Bp, long long bB, int ldb,
                        int Cin, int IH, int IW, int OW, const float* subMean,
                        const float* bias, float alpha, int relu,
                        const float* resid, long long rB,
                        float* Y, long long yB, int ldy, int M, int N, int K, int batch)
{
    dim3 grid((N + BN - 1) / BN, (M + BM - 1) / BM, batch);
    gemm_wmma_kernel<TA, TB, C3, AS><<<grid, dim3(TPB), 0, s>>>(
        A, aB, lda, Bp, bB, ldb, Cin, IH, IW, OW, subMean,
        bias, alpha, relu, resid, rB, Y, yB, ldy, M, N, K);
}

extern "C" void kernel_launch(void* const* d_in, const int* in_sizes, int n_in,
                              void* d_out, int out_size, void* d_ws, size_t ws_size,
                              hipStream_t stream)
{
    (void)in_sizes; (void)n_in; (void)out_size; (void)ws_size;

    const float* content = (const float*)d_in[0];
    const float* style   = (const float*)d_in[1];
    const float* wf_san = (const float*)d_in[2];  const float* bf_san = (const float*)d_in[3];
    const float* wg_san = (const float*)d_in[4];  const float* bg_san = (const float*)d_in[5];
    const float* wh_san = (const float*)d_in[6];  const float* bh_san = (const float*)d_in[7];
    const float* wf_mcc = (const float*)d_in[8];  const float* bf_mcc = (const float*)d_in[9];
    const float* wg_mcc = (const float*)d_in[10]; const float* bg_mcc = (const float*)d_in[11];
    const float* w_fc   = (const float*)d_in[12]; const float* b_fc   = (const float*)d_in[13];
    const float* w_out  = (const float*)d_in[14]; const float* b_out  = (const float*)d_in[15];
    const float* wc1 = (const float*)d_in[16]; const float* bc1 = (const float*)d_in[17];
    const float* wc2 = (const float*)d_in[18]; const float* bc2 = (const float*)d_in[19];
    const float* wc3 = (const float*)d_in[20]; const float* bc3 = (const float*)d_in[21];
    const float* ws1 = (const float*)d_in[22]; const float* bs1 = (const float*)d_in[23];
    const float* ws2 = (const float*)d_in[24]; const float* bs2 = (const float*)d_in[25];
    const float* ws3 = (const float*)d_in[26]; const float* bs3 = (const float*)d_in[27];
    const float* w_unc = (const float*)d_in[28]; const float* b_unc = (const float*)d_in[29];

    float* out = (float*)d_out;
    float* ws  = (float*)d_ws;

    const int Bn = 16, C = 512, HW = 1024;
    const long long NC  = (long long)C * HW;      // per-image plane stride
    const size_t NBHW   = (size_t)Bn * C * HW;    // 8,388,608 floats

    // workspace arena (slot reuse; peak ~302 MB)
    float* cn   = ws;                  // mvn(content)
    float* sn   = ws + 1 * NBHW;       // mvn(style)
    float* w1   = ws + 2 * NBHW;       // F_s -> g -> p
    float* gsb  = ws + 3 * NBHW;       // G_s -> F_Fc
    float* hsb  = ws + 4 * NBHW;       // H_s -> part2
    float* p3   = ws + 5 * NBHW;       // part3
    float* Sb   = ws + 6 * NBHW;       // S logits (2*NBHW), later phase-2 temps
    float* t1   = Sb;                            // [16,256,1024] / s1 [16,256,900]
    float* t2   = t1 + (size_t)Bn * 256 * HW;    // [16,128,1024] / s2 [16,128,784]
    float* fcb  = t2 + (size_t)Bn * 128 * HW;    // fc_ [16,32,1024]
    float* fsb  = fcb + (size_t)Bn * 32 * HW;    // fs  [16,32,784]
    float* wct  = fsb + (size_t)Bn * 32 * 784;   // [16,32,1024]
    float* cov  = wct + (size_t)Bn * 32 * HW;    // [16,32,32]
    float* smalls = ws + 8 * NBHW;
    float* mean_s = smalls;            // [B,C]
    float* fcs0   = smalls + 8192;     // [C,B]
    float* fcs    = smalls + 16384;    // [C,B]
    unsigned short* Sb16 = (unsigned short*)(smalls + 32768);  // bf16 softmax(S)

    // 1) MVN of content and style (+ spatial means of style for Fs0)
    mvn_kernel<<<Bn * C, 256, 0, stream>>>(content, cn, nullptr, HW, 1e-5f);
    mvn_kernel<<<Bn * C, 256, 0, stream>>>(style,   sn, mean_s,  HW, 1e-5f);

    // ---- SANet attention branch ----
    launch_gemm<0,0,0,0>(stream, wf_san, 0, C, cn, NC, HW, 0,0,0,0, nullptr,
                bf_san, 1.f, 0, nullptr, 0, w1,  NC, HW, C, HW, C, Bn);      // F_s
    launch_gemm<0,0,0,0>(stream, wg_san, 0, C, sn, NC, HW, 0,0,0,0, nullptr,
                bg_san, 1.f, 0, nullptr, 0, gsb, NC, HW, C, HW, C, Bn);      // G_s
    launch_gemm<0,0,0,0>(stream, wh_san, 0, C, style, NC, HW, 0,0,0,0, nullptr,
                bh_san, 1.f, 0, nullptr, 0, hsb, NC, HW, C, HW, C, Bn);      // H_s
    launch_gemm<1,0,0,0>(stream, w1, NC, HW, gsb, NC, HW, 0,0,0,0, nullptr,
                nullptr, 1.f, 0, nullptr, 0, Sb, (long long)HW * HW, HW,
                HW, HW, C, Bn);                                              // logits = F^T G
    softmax_kernel<<<Bn * HW, 256, 0, stream>>>(Sb, Sb16, HW);               // softmax -> bf16
    launch_gemm<0,0,0,1>(stream, hsb, NC, HW, (const float*)Sb16, (long long)HW * HW, HW,
                0,0,0,0, nullptr, nullptr, 1.f, 0, nullptr, 0,
                p3, NC, HW, C, HW, HW, Bn);                                  // part3 (async B)

    // ---- MCC branch ----
    launch_gemm<0,0,0,0>(stream, wf_mcc, 0, C, cn, NC, HW, 0,0,0,0, nullptr,
                bf_mcc, 1.f, 0, nullptr, 0, gsb, NC, HW, C, HW, C, Bn);      // F_Fc
    launch_gemm<0,0,0,0>(stream, wg_mcc, 0, C, sn, NC, HW, 0,0,0,0, nullptr,
                bg_mcc, 1.f, 0, nullptr, 0, w1,  NC, HW, C, HW, C, Bn);      // g
    mcc_stat_kernel<<<Bn * C, 256, 0, stream>>>(w1, fcs0, Bn, C, HW);        // sum g^2/sum g
    launch_gemm<0,0,0,0>(stream, w_fc, 0, C, fcs0, 0, Bn, 0,0,0,0, nullptr,
                b_fc, 1.f, 0, nullptr, 0, fcs, 0, Bn, C, Bn, C, 1);          // Linear

    // ---- WCT covariance branch ----
    launch_gemm<0,0,0,0>(stream, wc1, 0, 512, cn, NC, HW, 0,0,0,0, nullptr,
                bc1, 1.f, 1, nullptr, 0, t1, (long long)256 * HW, HW, 256, HW, 512, Bn);
    launch_gemm<0,0,0,0>(stream, wc2, 0, 256, t1, (long long)256 * HW, HW, 0,0,0,0, nullptr,
                bc2, 1.f, 1, nullptr, 0, t2, (long long)128 * HW, HW, 128, HW, 256, Bn);
    launch_gemm<0,0,0,0>(stream, wc3, 0, 128, t2, (long long)128 * HW, HW, 0,0,0,0, nullptr,
                bc3, 1.f, 0, nullptr, 0, fcb, (long long)32 * HW, HW, 32, HW, 128, Bn);
    // conv3x3 chain on (style - mean), VALID: 32->30->28
    launch_gemm<0,0,1,0>(stream, ws1, 0, 512 * 9, style, NC, 0,
                512, 32, 32, 30, mean_s,
                bs1, 1.f, 1, nullptr, 0, t1, (long long)256 * 900, 900,
                256, 900, 512 * 9, Bn);
    launch_gemm<0,0,1,0>(stream, ws2, 0, 256 * 9, t1, (long long)256 * 900, 0,
                256, 30, 30, 28, nullptr,
                bs2, 1.f, 1, nullptr, 0, t2, (long long)128 * 784, 784,
                128, 784, 256 * 9, Bn);
    launch_gemm<0,0,0,0>(stream, ws3, 0, 128, t2, (long long)128 * 784, 784, 0,0,0,0, nullptr,
                bs3, 1.f, 0, nullptr, 0, fsb, (long long)32 * 784, 784, 32, 784, 128, Bn);
    launch_gemm<0,1,0,0>(stream, fsb, (long long)32 * 784, 784, fsb, (long long)32 * 784, 784,
                0,0,0,0, nullptr, nullptr, 1.f / 784.f, 0, nullptr, 0,
                cov, 1024, 32, 32, 32, 784, Bn);                             // cov = fs fs^T
    launch_gemm<0,0,0,0>(stream, cov, 1024, 32, fcb, (long long)32 * HW, HW, 0,0,0,0, nullptr,
                nullptr, 1.f, 0, nullptr, 0, wct, (long long)32 * HW, HW, 32, HW, 32, Bn);
    launch_gemm<0,0,0,0>(stream, w_unc, 0, 32, wct, (long long)32 * HW, HW, 0,0,0,0, nullptr,
                b_unc, 1.f, 0, nullptr, 0, hsb, NC, HW, C, HW, 32, Bn);      // part2

    // ---- combine + output projection (+ residual content) ----
    const long long total = (long long)NBHW;
    combine_kernel<<<(unsigned)((total + 255) / 256), 256, 0, stream>>>(
        gsb, fcs, hsb, p3, w1, Bn, C, HW, total);                            // p
    launch_gemm<0,0,0,0>(stream, w_out, 0, C, w1, NC, HW, 0,0,0,0, nullptr,
                b_out, 1.f, 0, content, NC, out, NC, HW, C, HW, C, Bn);      // final
}
